// Llama3Attention_streaming_47923245089256
// MI455X (gfx1250) — matrix operand order repaired and checked by reading, not compiled, not run on hardware
//
#include <hip/hip_runtime.h>
#include <hip/hip_bf16.h>
#include <math.h>

#define S_LEN 2048
#define HID   4096
#define NH    32
#define KVH   8
#define HD    128
#define RECENT 204
#define SINKN  4
#define RROWS  208   // SINKN + RECENT

typedef __attribute__((ext_vector_type(8)))  float  v8f;
typedef __attribute__((ext_vector_type(16))) __bf16 v16bf;

union Frag32B { v16bf v; uint4 q[2]; };

__device__ __forceinline__ unsigned short f2bf(float f) {
    union { float f; unsigned u; } x; x.f = f;
    unsigned r = x.u + 0x7fffu + ((x.u >> 16) & 1u);
    return (unsigned short)(r >> 16);
}

// Async DMA: global -> LDS, 16 bytes per lane, tracked by ASYNCcnt.
// LDS byte address = low 32 bits of the generic pointer (ISA aperture rule).
__device__ __forceinline__ void async_b128(const void* g, void* lds) {
    unsigned loff = (unsigned)(uintptr_t)lds;
    unsigned long long ga = (unsigned long long)(uintptr_t)g;
    asm volatile("global_load_async_to_lds_b128 %0, %1, off"
                 :: "v"(loff), "v"(ga) : "memory");
}

// ---------------------------------------------------------------------------
// fp32 -> bf16 bulk convert (vectorized, n multiple of 4)
// ---------------------------------------------------------------------------
__global__ void cvt_bf16(const float* __restrict__ in,
                         unsigned short* __restrict__ out, int n) {
    int i = (blockIdx.x * blockDim.x + threadIdx.x) * 4;
    if (i >= n) return;
    float4 f = *(const float4*)(in + i);
    ushort4 o;
    o.x = f2bf(f.x); o.y = f2bf(f.y); o.z = f2bf(f.z); o.w = f2bf(f.w);
    *(ushort4*)(out + i) = o;
}

// ---------------------------------------------------------------------------
// C[M,N](f32) = A[M,K](bf16 row-major) * Bw[N,K](bf16 row-major)^T
// WG tile 128x128, 8 waves (4x2), wave tile 32x64, K-step 32.
// Double-buffered LDS panels filled by global_load_async_to_lds_b128,
// pipelined with s_wait_asynccnt; compute via v_wmma_f32_16x16x32_bf16.
// ---------------------------------------------------------------------------
__global__ __launch_bounds__(256)
void gemm_nt_bf16(const unsigned short* __restrict__ A,
                  const unsigned short* __restrict__ Bw,
                  float* __restrict__ C, int N, int K) {
    __shared__ unsigned short lA[2][128 * 40]; // 32 K elems + pad per row
    __shared__ unsigned short lB[2][128 * 40];

    const int tid     = threadIdx.x;
    const int wid     = tid >> 5;
    const int lane    = tid & 31;
    const int laneLow = lane & 15;
    const int hi      = lane >> 4;
    const int row0    = blockIdx.y * 128;
    const int col0    = blockIdx.x * 128;
    const int waveM   = (wid & 3) * 32;
    const int waveN   = (wid >> 2) * 64;

    v8f acc[2][4];
#pragma unroll
    for (int mt = 0; mt < 2; ++mt)
#pragma unroll
        for (int nt = 0; nt < 4; ++nt)
#pragma unroll
            for (int r = 0; r < 8; ++r) acc[mt][nt][r] = 0.0f;

    // cooperative loader mapping: each thread stages 2x16B of A and of B
    const int lrow = tid >> 1;            // 0..127
    const int lseg = (tid & 1) * 16;      // element offset 0 or 16
    const unsigned short* aSrc = A  + (size_t)(row0 + lrow) * K + lseg;
    const unsigned short* bSrc = Bw + (size_t)(col0 + lrow) * K + lseg;
    const int nk = K >> 5;

    auto stage = [&](int kt, int buf) {
        const unsigned short* ap = aSrc + kt * 32;
        const unsigned short* bp = bSrc + kt * 32;
        unsigned short* la = &lA[buf][lrow * 40 + lseg];
        unsigned short* lb = &lB[buf][lrow * 40 + lseg];
        async_b128(ap,     la);
        async_b128(ap + 8, la + 8);
        async_b128(bp,     lb);
        async_b128(bp + 8, lb + 8);
    };

    auto compute = [&](int buf) {
        Frag32B af[2];
#pragma unroll
        for (int mt = 0; mt < 2; ++mt) {
            const int r = waveM + mt * 16 + laneLow;
            af[mt].q[0] = *(const uint4*)(&lA[buf][r * 40 + 8 * hi]);
            af[mt].q[1] = *(const uint4*)(&lA[buf][r * 40 + 16 + 8 * hi]);
        }
#pragma unroll
        for (int nt = 0; nt < 4; ++nt) {
            const int r = waveN + nt * 16 + laneLow;
            Frag32B bf_;
            bf_.q[0] = *(const uint4*)(&lB[buf][r * 40 + 8 * hi]);
            bf_.q[1] = *(const uint4*)(&lB[buf][r * 40 + 16 + 8 * hi]);
            acc[0][nt] = __builtin_amdgcn_wmma_f32_16x16x32_bf16(
                false, af[0].v, false, bf_.v, (short)0, acc[0][nt], false, false);
            acc[1][nt] = __builtin_amdgcn_wmma_f32_16x16x32_bf16(
                false, af[1].v, false, bf_.v, (short)0, acc[1][nt], false, false);
        }
    };

    stage(0, 0);
    for (int kt = 0; kt < nk - 1; ++kt) {
        stage(kt + 1, (kt + 1) & 1);               // prefetch next panel
        asm volatile("s_wait_asynccnt 0x4" ::: "memory"); // current panel done
        __syncthreads();
        compute(kt & 1);
        __syncthreads();                           // all reads done before reuse
    }
    asm volatile("s_wait_asynccnt 0x0" ::: "memory");
    __syncthreads();
    compute((nk - 1) & 1);

#pragma unroll
    for (int mt = 0; mt < 2; ++mt)
#pragma unroll
        for (int nt = 0; nt < 4; ++nt) {
            const int col  = col0 + waveN + nt * 16 + laneLow;
            const int rowb = row0 + waveM + mt * 16 + 8 * hi;
#pragma unroll
            for (int r = 0; r < 8; ++r)
                C[(size_t)(rowb + r) * N + col] = acc[mt][nt][r];
        }
}

// ---------------------------------------------------------------------------
// RoPE on Q (f32 -> bf16 row-major, ready as WMMA A operand)
// ---------------------------------------------------------------------------
__global__ void rope_q(const float* __restrict__ qf, const int* __restrict__ pos,
                       unsigned short* __restrict__ qb) {
    int t = blockIdx.x * blockDim.x + threadIdx.x;
    if (t >= S_LEN * NH * 64) return;
    const int d2 = t & 63;
    const int h  = (t >> 6) & 31;
    const int s  = t >> 11;
    const float p    = (float)pos[s];
    const float freq = __expf(-(float)d2 * (13.122363377404328f / 64.0f));
    const float ang  = p * freq;
    const float c = __cosf(ang), sn = __sinf(ang);
    const size_t base = (size_t)s * HID + h * HD + d2;
    const float q0 = qf[base], q1 = qf[base + 64];
    qb[base]      = f2bf(q0 * c - q1 * sn);
    qb[base + 64] = f2bf(q1 * c + q0 * sn);
}

// ---------------------------------------------------------------------------
// RoPE on K + pack K (bf16 [g][s][d]) and V transposed (bf16 [g][d][s]),
// and emit the f32 sink+recent K/V slices straight into d_out.
// ---------------------------------------------------------------------------
__global__ void pack_kv(const float* __restrict__ kf, const float* __restrict__ vf,
                        const int* __restrict__ pos,
                        unsigned short* __restrict__ kb,
                        unsigned short* __restrict__ vtb,
                        float* __restrict__ krec, float* __restrict__ vrec) {
    int t = blockIdx.x * blockDim.x + threadIdx.x;
    if (t >= S_LEN * KVH * 64) return;
    const int d2 = t & 63;
    const int g  = (t >> 6) & 7;
    const int s  = t >> 9;
    const float p    = (float)pos[s];
    const float freq = __expf(-(float)d2 * (13.122363377404328f / 64.0f));
    const float ang  = p * freq;
    const float c = __cosf(ang), sn = __sinf(ang);

    const size_t base = (size_t)s * (KVH * HD) + g * HD + d2;
    const float k0 = kf[base], k1 = kf[base + 64];
    const float kr0 = k0 * c - k1 * sn;
    const float kr1 = k1 * c + k0 * sn;
    const size_t kbi = ((size_t)g * S_LEN + s) * HD + d2;
    kb[kbi]      = f2bf(kr0);
    kb[kbi + 64] = f2bf(kr1);

    const float v0 = vf[base], v1 = vf[base + 64];
    vtb[((size_t)g * HD + d2) * S_LEN + s]      = f2bf(v0);
    vtb[((size_t)g * HD + d2 + 64) * S_LEN + s] = f2bf(v1);

    int row = -1;
    if (s < SINKN)                row = s;
    else if (s >= S_LEN - RECENT) row = s - (S_LEN - RECENT) + SINKN;
    if (row >= 0) {
        const size_t ri = ((size_t)g * RROWS + row) * HD + d2;
        krec[ri] = kr0; krec[ri + 64] = kr1;
        vrec[ri] = v0;  vrec[ri + 64] = v1;
    }
}

// ---------------------------------------------------------------------------
// Flash attention: one workgroup = (128 q rows, one head). 8 waves, each owns
// 16 q rows and the full D=128. K/V^T blocks async-DMAed into double-buffered
// LDS; scores + PV via v_wmma_f32_16x16x32_bf16 with online softmax.
// ---------------------------------------------------------------------------
__global__ __launch_bounds__(256)
void attn_flash(const unsigned short* __restrict__ qb,
                const unsigned short* __restrict__ kb,
                const unsigned short* __restrict__ vtb,
                unsigned short* __restrict__ attnb) {
    __shared__ unsigned short kt_s[2][32 * 136];   // [key][d], padded row
    __shared__ unsigned short vt_s[2][128 * 40];   // [d][key], padded row
    __shared__ unsigned short p_s[8 * 16 * 40];    // per-wave P scratch

    const int qblock  = blockIdx.x;
    const int h       = blockIdx.y;
    const int g       = h >> 2;          // GQA group (n_rep = 4)
    const int tid     = threadIdx.x;
    const int wid     = tid >> 5;
    const int lane    = tid & 31;
    const int laneLow = lane & 15;
    const int hi      = lane >> 4;
    const int qr0     = qblock * 128 + wid * 16;

    // Q fragments for this wave: 16 rows x 128 d as 4 A-operands
    Frag32B qfr[4];
    {
        const unsigned short* qp = qb + (size_t)(qr0 + laneLow) * HID + h * HD;
#pragma unroll
        for (int ks = 0; ks < 4; ++ks) {
            qfr[ks].q[0] = *(const uint4*)(qp + ks * 32 + 8 * hi);
            qfr[ks].q[1] = *(const uint4*)(qp + ks * 32 + 16 + 8 * hi);
        }
    }

    v8f acc[8];
#pragma unroll
    for (int nt = 0; nt < 8; ++nt)
#pragma unroll
        for (int r = 0; r < 8; ++r) acc[nt][r] = 0.0f;
    float m[8], l[8];
#pragma unroll
    for (int r = 0; r < 8; ++r) { m[r] = -3.0e38f; l[r] = 0.0f; }

    const float scale = 0.08838834764831845f;  // 1/sqrt(128)
    const int nblk = qblock * 4 + 4;
    const unsigned short* kg = kb  + (size_t)g * S_LEN * HD;
    const unsigned short* vg = vtb + (size_t)g * HD * S_LEN;
    unsigned short* pw = p_s + wid * 16 * 40;

    auto stageKV = [&](int blk, int buf) {
        const int key0 = blk * 32;
#pragma unroll
        for (int j = 0; j < 2; ++j) {
            const int i  = tid * 2 + j;
            const int kr = i >> 4, ksg = (i & 15) * 8;
            async_b128(kg + (size_t)(key0 + kr) * HD + ksg,
                       &kt_s[buf][kr * 136 + ksg]);
            const int vr = i >> 2, vsg = (i & 3) * 8;
            async_b128(vg + (size_t)vr * S_LEN + key0 + vsg,
                       &vt_s[buf][vr * 40 + vsg]);
        }
    };

    stageKV(0, 0);
    for (int blk = 0; blk < nblk; ++blk) {
        if (blk + 1 < nblk) {
            stageKV(blk + 1, (blk + 1) & 1);      // prefetch next K/V block
            asm volatile("s_wait_asynccnt 0x4" ::: "memory");
        } else {
            asm volatile("s_wait_asynccnt 0x0" ::: "memory");
        }
        __syncthreads();

        const int key0 = blk * 32;
        const int buf  = blk & 1;
        if (key0 <= qr0 + 15) {                   // wave-uniform causal bound
            // scores: two 16x16 tiles (keys key0..+15, key0+16..+31)
            v8f sc[2];
#pragma unroll
            for (int t = 0; t < 2; ++t) {
                v8f s = {0.f, 0.f, 0.f, 0.f, 0.f, 0.f, 0.f, 0.f};
#pragma unroll
                for (int ks = 0; ks < 4; ++ks) {
                    Frag32B kf_;
                    const int r = t * 16 + laneLow;
                    kf_.q[0] = *(const uint4*)(&kt_s[buf][r * 136 + ks * 32 + 8 * hi]);
                    kf_.q[1] = *(const uint4*)(&kt_s[buf][r * 136 + ks * 32 + 16 + 8 * hi]);
                    s = __builtin_amdgcn_wmma_f32_16x16x32_bf16(
                        false, qfr[ks].v, false, kf_.v, (short)0, s, false, false);
                }
                sc[t] = s;
            }
            // scale + causal mask
#pragma unroll
            for (int t = 0; t < 2; ++t)
#pragma unroll
                for (int r = 0; r < 8; ++r) {
                    const int kcol = key0 + t * 16 + laneLow;
                    const int qrow = qr0 + 8 * hi + r;
                    const float v  = sc[t][r] * scale;
                    sc[t][r] = (kcol > qrow) ? -3.0e38f : v;
                }

            // online softmax (row reductions across the 16-lane half-wave)
            float p0[8], p1[8];
#pragma unroll
            for (int r = 0; r < 8; ++r) {
                float bmax = fmaxf(sc[0][r], sc[1][r]);
#pragma unroll
                for (int off = 8; off >= 1; off >>= 1)
                    bmax = fmaxf(bmax, __shfl_xor(bmax, off, 32));
                const float mn    = fmaxf(m[r], bmax);
                const float alpha = __expf(m[r] - mn);
                const float e0 = __expf(sc[0][r] - mn);
                const float e1 = __expf(sc[1][r] - mn);
                float rs = e0 + e1;
#pragma unroll
                for (int off = 8; off >= 1; off >>= 1)
                    rs += __shfl_xor(rs, off, 32);
                l[r] = l[r] * alpha + rs;
                m[r] = mn;
                p0[r] = e0; p1[r] = e1;
#pragma unroll
                for (int nt = 0; nt < 8; ++nt) acc[nt][r] *= alpha;
            }

            // P: C-layout -> A-operand layout via wave-private LDS (bf16)
#pragma unroll
            for (int r = 0; r < 8; ++r) {
                pw[(8 * hi + r) * 40 + laneLow]      = f2bf(p0[r]);
                pw[(8 * hi + r) * 40 + 16 + laneLow] = f2bf(p1[r]);
            }
            asm volatile("s_wait_dscnt 0" ::: "memory");
            Frag32B pf;
            pf.q[0] = *(const uint4*)(pw + laneLow * 40 + 8 * hi);
            pf.q[1] = *(const uint4*)(pw + laneLow * 40 + 16 + 8 * hi);

            // P (16x32) x V (32x128): 8 WMMAs
#pragma unroll
            for (int nt = 0; nt < 8; ++nt) {
                Frag32B vf_;
                const int r = nt * 16 + laneLow;
                vf_.q[0] = *(const uint4*)(&vt_s[buf][r * 40 + 8 * hi]);
                vf_.q[1] = *(const uint4*)(&vt_s[buf][r * 40 + 16 + 8 * hi]);
                acc[nt] = __builtin_amdgcn_wmma_f32_16x16x32_bf16(
                    false, pf.v, false, vf_.v, (short)0, acc[nt], false, false);
            }
        }
        __syncthreads();   // all reads of this buffer done before it is refilled
    }

    // epilogue: normalize, store bf16 row-major [s][h*D+d]
#pragma unroll
    for (int r = 0; r < 8; ++r) {
        const float inv = 1.0f / l[r];
        const int row = qr0 + 8 * hi + r;
#pragma unroll
        for (int nt = 0; nt < 8; ++nt)
            attnb[(size_t)row * HID + h * HD + nt * 16 + laneLow] =
                f2bf(acc[nt][r] * inv);
    }
}

// ---------------------------------------------------------------------------
extern "C" void kernel_launch(void* const* d_in, const int* in_sizes, int n_in,
                              void* d_out, int out_size, void* d_ws, size_t ws_size,
                              hipStream_t stream) {
    (void)in_sizes; (void)n_in; (void)out_size; (void)ws_size;
    const float* hs  = (const float*)d_in[0];
    const int*   pos = (const int*)d_in[1];
    const float* Wq  = (const float*)d_in[2];
    const float* Wk  = (const float*)d_in[3];
    const float* Wv  = (const float*)d_in[4];
    const float* Wo  = (const float*)d_in[5];

    float* out  = (float*)d_out;
    float* krec = out + (size_t)S_LEN * HID;
    float* vrec = krec + (size_t)KVH * RROWS * HD;

    char* ws = (char*)d_ws;
    size_t off = 0;
    unsigned short* hsb = (unsigned short*)(ws + off); off += (size_t)S_LEN * HID * 2;
    unsigned short* wqb = (unsigned short*)(ws + off); off += (size_t)HID * HID * 2;
    unsigned short* wkb = (unsigned short*)(ws + off); off += (size_t)KVH * HD * HID * 2;
    unsigned short* wvb = (unsigned short*)(ws + off); off += (size_t)KVH * HD * HID * 2;
    unsigned short* wob = (unsigned short*)(ws + off); off += (size_t)HID * HID * 2;
    float* qf = (float*)(ws + off); off += (size_t)S_LEN * HID * 4;
    float* kf = (float*)(ws + off); off += (size_t)S_LEN * KVH * HD * 4;
    float* vf = (float*)(ws + off); off += (size_t)S_LEN * KVH * HD * 4;
    unsigned short* qbp  = (unsigned short*)(ws + off); off += (size_t)S_LEN * HID * 2;
    unsigned short* kbp  = (unsigned short*)(ws + off); off += (size_t)KVH * S_LEN * HD * 2;
    unsigned short* vtbp = (unsigned short*)(ws + off); off += (size_t)KVH * HD * S_LEN * 2;
    unsigned short* attb = (unsigned short*)(ws + off); off += (size_t)S_LEN * HID * 2;

    // 1) fp32 -> bf16 conversions
    cvt_bf16<<<(S_LEN * HID / 4 + 255) / 256, 256, 0, stream>>>(hs, hsb, S_LEN * HID);
    cvt_bf16<<<(HID * HID / 4 + 255) / 256, 256, 0, stream>>>(Wq, wqb, HID * HID);
    cvt_bf16<<<(KVH * HD * HID / 4 + 255) / 256, 256, 0, stream>>>(Wk, wkb, KVH * HD * HID);
    cvt_bf16<<<(KVH * HD * HID / 4 + 255) / 256, 256, 0, stream>>>(Wv, wvb, KVH * HD * HID);
    cvt_bf16<<<(HID * HID / 4 + 255) / 256, 256, 0, stream>>>(Wo, wob, HID * HID);

    // 2) QKV projections (WMMA bf16 GEMM, async-LDS pipeline)
    gemm_nt_bf16<<<dim3(HID / 128, S_LEN / 128), 256, 0, stream>>>(hsb, wqb, qf, HID, HID);
    gemm_nt_bf16<<<dim3(KVH * HD / 128, S_LEN / 128), 256, 0, stream>>>(hsb, wkb, kf, KVH * HD, HID);
    gemm_nt_bf16<<<dim3(KVH * HD / 128, S_LEN / 128), 256, 0, stream>>>(hsb, wvb, vf, KVH * HD, HID);

    // 3) RoPE + packing (+ streaming K/V slices into d_out)
    rope_q<<<S_LEN * NH * 64 / 256, 256, 0, stream>>>(qf, pos, qbp);
    pack_kv<<<S_LEN * KVH * 64 / 256, 256, 0, stream>>>(kf, vf, pos, kbp, vtbp, krec, vrec);

    // 4) flash attention (WMMA scores + PV, async-LDS double buffering)
    attn_flash<<<dim3(S_LEN / 128, NH), 256, 0, stream>>>(qbp, kbp, vtbp, attb);

    // 5) output projection -> d_out
    gemm_nt_bf16<<<dim3(HID / 128, S_LEN / 128), 256, 0, stream>>>(attb, wob, out, HID, HID);
}
